// AttentionNet_80925773791316
// MI455X (gfx1250) — matrix operand order repaired and checked by reading, not compiled, not run on hardware
//
#include <hip/hip_runtime.h>

typedef __attribute__((ext_vector_type(16))) _Float16 v16h;
typedef __attribute__((ext_vector_type(8)))  _Float16 v8h;
typedef __attribute__((ext_vector_type(8)))  float    v8f;

#define TT   200   // sequence length
#define HH   64    // head dim
#define HID1 80    // first hidden
#define HID2 40    // second hidden
#define NT1  5     // N tiles for GEMM1 (80/16)
#define NW   13    // waves per block (ceil(200/16))
#define TPAD 208   // padded T (13*16)
#define NTHREADS (NW * 32)

union F16x16 { v16h v; v8h h[2]; };

__global__ __launch_bounds__(NTHREADS)
void din_attn_kernel(const float* __restrict__ query,
                     const float* __restrict__ keys,
                     const int*   __restrict__ keys_length,
                     const float* __restrict__ W1,  const float* __restrict__ b1,
                     const float* __restrict__ W2,  const float* __restrict__ b2,
                     const float* __restrict__ Wfc, const float* __restrict__ bfc,
                     float* __restrict__ out)
{
    // B-fragment-swizzled weights: [kstep][ntile][lane][16 contiguous halves]
    __shared__ __align__(32) _Float16 s_W1f[8][NT1][32][16];   // 40960 B
    __shared__ __align__(32) _Float16 s_W2f[3][3][32][16];     //  9216 B
    __shared__ __align__(32) _Float16 s_x1[NW][16][32];        // 13312 B
    __shared__ __align__(16) _Float16 s_qh[HH];                //   128 B
    __shared__ float s_scores[TPAD];                           //   832 B
    __shared__ float s_red[16];                                //    64 B
    // s_part (6*64 floats = 1536 B) overlays s_x1: disjoint, barrier-separated phases
    float* s_part = (float*)&s_x1[0][0][0];

    const int tid  = threadIdx.x;
    const int lane = tid & 31;
    const int wid  = tid >> 5;          // 0..12
    const int lo   = lane & 15;
    const int hi   = lane >> 4;
    const int b    = blockIdx.x;

    // ---- one-time block staging: W1/W2 -> f16 fragments in LDS -------
    for (int e = tid; e < 256 * HID1; e += NTHREADS) {
        int k = e / HID1, c = e - k * HID1;
        // B layout: element j of lane (lo,hi) is W1[kk*32 + 16*hi + j][16n + lo]
        s_W1f[k >> 5][c >> 4][(((k >> 4) & 1) << 4) | (c & 15)][k & 15] =
            (_Float16)W1[e];
    }
    for (int idx = tid; idx < 3 * 3 * 32 * 16; idx += NTHREADS) {
        int j = idx & 15, ln = (idx >> 4) & 31, rem = idx >> 9;
        int n = rem % 3, kc = rem / 3;
        int k = kc * 32 + ((ln >> 4) << 4) + j;
        int c = n * 16 + (ln & 15);
        s_W2f[kc][n][ln][j] = (k < HID1 && c < HID2) ? (_Float16)W2[k * HID2 + c]
                                                     : (_Float16)0.f;
    }
    if (tid < HH) s_qh[tid] = (_Float16)query[b * HH + tid];
    __syncthreads();

    const int t0   = wid * 16;
    const int L    = keys_length[b];
    const int trow = t0 + lo;
    const float* keyrow =
        keys + ((size_t)b * TT + (trow < TT ? trow : (TT - 1))) * HH;

    // ---- per-lane: load this row's keys once as 4x v8h, q as 4x v8h --
    // run i: columns [ (i>>1)*32 + (i&1)*8 + 16*hi , +8 )
    v8h kh[4], qh4[4];
    #pragma unroll
    for (int i = 0; i < 4; ++i) {
        const int base = ((i >> 1) << 5) + ((i & 1) << 3) + (hi << 4);
        v8h t;
        #pragma unroll
        for (int u = 0; u < 8; ++u) t[u] = (_Float16)keyrow[base + u];
        kh[i]  = t;
        qh4[i] = *(const v8h*)&s_qh[base];
    }

    const v8f vzero = {0.f, 0.f, 0.f, 0.f, 0.f, 0.f, 0.f, 0.f};

    // ---------------- GEMM1: din[16x256] @ W1[256x80] ----------------
    v8f acc1[NT1];
    #pragma unroll
    for (int n = 0; n < NT1; ++n) acc1[n] = vzero;

    #pragma unroll
    for (int kk = 0; kk < 8; ++kk) {
        const int seg = kk >> 1;   // 0:q 1:k 2:q-k 3:q*k
        const int ci  = kk & 1;    // which 32-col half of the 64-col segment
        F16x16 a;
        #pragma unroll
        for (int r = 0; r < 2; ++r) {
            v8h q8 = qh4[(ci << 1) | r];
            v8h k8 = kh [(ci << 1) | r];
            a.h[r] = (seg == 0) ? q8 : (seg == 1) ? k8
                   : (seg == 2) ? (q8 - k8) : (q8 * k8);
        }
        #pragma unroll
        for (int n = 0; n < NT1; ++n) {
            v16h bm = *(const v16h*)&s_W1f[kk][n][lane][0];
            acc1[n] = __builtin_amdgcn_wmma_f32_16x16x32_f16(
                          false, a.v, false, bm, (short)0, acc1[n], false, false);
        }
    }

    // bias + relu ; lane holds x1[m = v+8*hi][c = 16n+lo]
    float x1v[NT1][8];
    #pragma unroll
    for (int n = 0; n < NT1; ++n) {
        float bb = b1[n * 16 + lo];
        #pragma unroll
        for (int v = 0; v < 8; ++v)
            x1v[n][v] = fmaxf(acc1[n][v] + bb, 0.f);
    }

    // ---------------- GEMM2: x1[16x80] @ W2[80x40], K padded to 96 ----
    v8f acc2[3];
    #pragma unroll
    for (int n = 0; n < 3; ++n) acc2[n] = vzero;

    for (int kc = 0; kc < 3; ++kc) {
        const int n0 = 2 * kc, n1 = 2 * kc + 1;
        #pragma unroll
        for (int v = 0; v < 8; ++v) {
            int m = v + 8 * hi;
            s_x1[wid][m][lo]      = (n0 < NT1) ? (_Float16)x1v[n0][v] : (_Float16)0.f;
            s_x1[wid][m][16 + lo] = (n1 < NT1) ? (_Float16)x1v[n1][v] : (_Float16)0.f;
        }
        __syncthreads();   // cross-lane transpose visibility

        F16x16 a;
        a.h[0] = *(const v8h*)&s_x1[wid][lo][hi << 4];
        a.h[1] = *(const v8h*)&s_x1[wid][lo][(hi << 4) + 8];
        #pragma unroll
        for (int n = 0; n < 3; ++n) {
            v16h bm = *(const v16h*)&s_W2f[kc][n][lane][0];
            acc2[n] = __builtin_amdgcn_wmma_f32_16x16x32_f16(
                          false, a.v, false, bm, (short)0, acc2[n], false, false);
        }
        __syncthreads();   // WAR on staging buffer before next chunk
    }

    // ---- bias + relu + dot with Wfc -> per-row score -----------------
    float p[8] = {0.f, 0.f, 0.f, 0.f, 0.f, 0.f, 0.f, 0.f};
    #pragma unroll
    for (int n = 0; n < 3; ++n) {
        int   cc = n * 16 + lo;
        float bb = (cc < HID2) ? b2[cc]  : 0.f;
        float wf = (cc < HID2) ? Wfc[cc] : 0.f;
        #pragma unroll
        for (int v = 0; v < 8; ++v) {
            float x2 = fmaxf(acc2[n][v] + bb, 0.f);
            p[v] += x2 * wf;
        }
    }
    #pragma unroll
    for (int v = 0; v < 8; ++v) {
        #pragma unroll
        for (int off = 1; off < 16; off <<= 1)
            p[v] += __shfl_xor(p[v], off, 16);   // reduce across 16 columns
    }
    const float bfc0 = bfc[0];
    if (lo == 0) {
        #pragma unroll
        for (int v = 0; v < 8; ++v) {
            int  t     = t0 + v + 8 * hi;
            bool valid = (t < TT) && (t < L);
            s_scores[t] = valid ? (p[v] + bfc0) * 0.125f : -INFINITY;
        }
    }
    __syncthreads();   // after this barrier s_x1 is dead -> s_part may reuse it

    // ---------------- softmax over T (block-wide) ---------------------
    float sv = (tid < TPAD) ? s_scores[tid] : -INFINITY;
    float mx = sv;
    #pragma unroll
    for (int off = 1; off < 32; off <<= 1)
        mx = fmaxf(mx, __shfl_xor(mx, off, 32));
    if (lane == 0) s_red[wid] = mx;
    __syncthreads();
    if (tid == 0) {
        float m = -INFINITY;
        for (int i = 0; i < NW; ++i) m = fmaxf(m, s_red[i]);
        s_red[14] = m;
    }
    __syncthreads();
    const float gm = s_red[14];
    float e = (tid < TPAD) ? __expf(sv - gm) : 0.f;
    if (tid < TPAD) s_scores[tid] = e;
    float sm = e;
    #pragma unroll
    for (int off = 1; off < 32; off <<= 1)
        sm += __shfl_xor(sm, off, 32);
    if (lane == 0) s_red[wid] = sm;
    __syncthreads();
    if (tid == 0) {
        float s = 0.f;
        for (int i = 0; i < NW; ++i) s += s_red[i];
        s_red[15] = s;
    }
    __syncthreads();
    const float inv = 1.0f / s_red[15];

    // ---------------- out[b,h] = sum_t attn[t] * keys[b,t,h] ----------
    if (tid < 6 * HH) {
        int   g = tid >> 6, h = tid & 63;
        float acc = 0.f;
        for (int t = g; t < TT; t += 6)
            acc += s_scores[t] * keys[((size_t)b * TT + t) * HH + h];
        s_part[tid] = acc;
    }
    __syncthreads();
    if (tid < HH) {
        float a = 0.f;
        #pragma unroll
        for (int g = 0; g < 6; ++g) a += s_part[g * HH + tid];
        out[b * HH + tid] = a * inv;
    }
}

extern "C" void kernel_launch(void* const* d_in, const int* in_sizes, int n_in,
                              void* d_out, int out_size, void* d_ws, size_t ws_size,
                              hipStream_t stream)
{
    const float* query = (const float*)d_in[0];
    const float* keys  = (const float*)d_in[1];
    const int*   klen  = (const int*)  d_in[2];
    const float* W1    = (const float*)d_in[3];
    const float* b1    = (const float*)d_in[4];
    const float* W2    = (const float*)d_in[5];
    const float* b2    = (const float*)d_in[6];
    const float* Wfc   = (const float*)d_in[7];
    const float* bfc   = (const float*)d_in[8];
    float* out = (float*)d_out;

    const int B = in_sizes[0] / HH;   // 2048
    din_attn_kernel<<<B, NTHREADS, 0, stream>>>(query, keys, klen,
                                                W1, b1, W2, b2, Wfc, bfc, out);
}
